// GptOssMLP_63307817943054
// MI455X (gfx1250) — compile-verified, matrix operand order checked by reading, not run on hardware
//
#include <hip/hip_runtime.h>
#include <hip/hip_bf16.h>
#include <math.h>

// ---------------------------------------------------------------------------
// GPT-OSS MoE MLP for MI455X (gfx1250, wave32, WMMA bf16, async-LDS staging)
// ---------------------------------------------------------------------------

typedef __attribute__((ext_vector_type(16))) __bf16 v16bf;
typedef __attribute__((ext_vector_type(8)))  __bf16 bf16x8;
typedef __attribute__((ext_vector_type(4)))  __bf16 bf16x4;
typedef __attribute__((ext_vector_type(8)))  float  v8f;
typedef __attribute__((ext_vector_type(4)))  float  f32x4;

#define ALPHA_C 1.702f
#define LIMIT_C 7.0f

constexpr int E      = 8;
constexpr int H      = 2048;
constexpr int I_DIM  = 2048;
constexpr int TWOI   = 4096;
constexpr int T      = 4096;      // B*S
constexpr int PAIRS  = 2 * T;     // every token routes to exactly 2 experts
constexpr int BK     = 32;        // K-step per WMMA sweep
constexpr int KPAD   = 40;        // LDS row pitch in halves (80B, 16B aligned)
constexpr int KSTEPS = 64;        // 2048 / 32 for both GEMMs

// ---- CDNA5 async global->LDS copy (GLOBAL_LOAD_ASYNC_TO_LDS_B128) ----------
#if defined(__has_builtin)
#  if __has_builtin(__builtin_amdgcn_global_load_async_to_lds_b128)
#    define HAVE_ASYNC_COPY 1
#  endif
#endif
#ifndef HAVE_ASYNC_COPY
#  define HAVE_ASYNC_COPY 0
#endif

#if HAVE_ASYNC_COPY
typedef int v4i_b128 __attribute__((vector_size(16)));   // matches builtin proto
__device__ __forceinline__ void async_copy16(const void* g, void* l) {
    // generic LDS pointer low 32 bits == LDS byte offset (ISA 10.2 aperture rule)
    __attribute__((address_space(1))) v4i_b128* gp =
        (__attribute__((address_space(1))) v4i_b128*)(uintptr_t)g;
    __attribute__((address_space(3))) v4i_b128* lp =
        (__attribute__((address_space(3))) v4i_b128*)(uintptr_t)(uint32_t)(uintptr_t)l;
    __builtin_amdgcn_global_load_async_to_lds_b128(gp, lp, 0, 0);
}
__device__ __forceinline__ void async_wait0() {
#  if __has_builtin(__builtin_amdgcn_s_wait_asynccnt)
    __builtin_amdgcn_s_wait_asynccnt(0);
#  else
    asm volatile("s_wait_asynccnt 0x0" ::: "memory");
#  endif
}
#endif

// ---------------------------------------------------------------------------
// fp32 -> bf16 conversion of hidden states (A matrix reused across N tiles)
// ---------------------------------------------------------------------------
__global__ void cvt_hidden_kernel(const float* __restrict__ h,
                                  __bf16* __restrict__ hb) {
    int i = (blockIdx.x * blockDim.x + threadIdx.x) * 4;
    f32x4 v = *(const f32x4*)(h + i);
    bf16x4 o;
    o[0] = (__bf16)v[0]; o[1] = (__bf16)v[1];
    o[2] = (__bf16)v[2]; o[3] = (__bf16)v[3];
    *(bf16x4*)(hb + i) = o;
}

// ---------------------------------------------------------------------------
// Router: one wave32 per token. 8 logits via shuffle reduction, top-2 softmax.
// ---------------------------------------------------------------------------
__global__ void router_kernel(const float* __restrict__ h,
                              const float* __restrict__ rw,
                              const float* __restrict__ rb,
                              int*  __restrict__ cnt,
                              int*  __restrict__ topidx,
                              float* __restrict__ topw) {
    const int wave = threadIdx.x >> 5;
    const int lane = threadIdx.x & 31;
    const int t    = blockIdx.x * (blockDim.x >> 5) + wave;
    if (t >= T) return;

    float acc[E];
#pragma unroll
    for (int e = 0; e < E; ++e) acc[e] = 0.0f;

    const float* hp = h + (size_t)t * H;
    for (int k = lane; k < H; k += 32) {
        float hv = hp[k];
#pragma unroll
        for (int e = 0; e < E; ++e) acc[e] += hv * rw[e * H + k];
    }
#pragma unroll
    for (int e = 0; e < E; ++e) {
#pragma unroll
        for (int s = 16; s > 0; s >>= 1) acc[e] += __shfl_xor(acc[e], s, 32);
    }
    if (lane == 0) {
        float lg[E];
#pragma unroll
        for (int e = 0; e < E; ++e) lg[e] = acc[e] + rb[e];
        int   i0 = 0; float v0 = lg[0];
#pragma unroll
        for (int e = 1; e < E; ++e) if (lg[e] > v0) { v0 = lg[e]; i0 = e; }
        int   i1 = (i0 == 0) ? 1 : 0; float v1 = lg[i1];
#pragma unroll
        for (int e = 0; e < E; ++e)
            if (e != i0 && lg[e] > v1) { v1 = lg[e]; i1 = e; }
        float e1  = __expf(v1 - v0);                 // e0 = exp(0) = 1
        float inv = __builtin_amdgcn_rcpf(1.0f + e1);
        topidx[2 * t + 0] = i0;  topw[2 * t + 0] = inv;
        topidx[2 * t + 1] = i1;  topw[2 * t + 1] = e1 * inv;
        atomicAdd(&cnt[i0], 1);
        atomicAdd(&cnt[i1], 1);
    }
}

// ---------------------------------------------------------------------------
// Exclusive scan over 8 expert counts (trivial) + reset fill counters
// ---------------------------------------------------------------------------
__global__ void offsets_kernel(const int* __restrict__ cnt,
                               int* __restrict__ off,
                               int* __restrict__ fill) {
    if (threadIdx.x == 0) {
        int s = 0;
#pragma unroll
        for (int e = 0; e < E; ++e) { off[e] = s; s += cnt[e]; }
        off[E] = s;
    }
    if (threadIdx.x < E) fill[threadIdx.x] = 0;
}

// ---------------------------------------------------------------------------
// Scatter token ids + routing weights into compacted per-expert lists
// ---------------------------------------------------------------------------
__global__ void scatter_kernel(const int* __restrict__ topidx,
                               const float* __restrict__ topw,
                               const int* __restrict__ off,
                               int* __restrict__ fill,
                               int* __restrict__ rows,
                               float* __restrict__ rscore) {
    int t = blockIdx.x * blockDim.x + threadIdx.x;
    if (t >= T) return;
#pragma unroll
    for (int s = 0; s < 2; ++s) {
        int e = topidx[2 * t + s];
        int p = atomicAdd(&fill[e], 1);
        int r = off[e] + p;
        rows[r]   = t;
        rscore[r] = topw[2 * t + s];
    }
}

// ---------------------------------------------------------------------------
// helper: build v16bf fragment from two 16B LDS reads (ISA 7.12.2 A/B layout)
// ---------------------------------------------------------------------------
__device__ __forceinline__ v16bf make_frag(const __bf16* base, int row, int kOff) {
    bf16x8 lo = *(const bf16x8*)(base + row * KPAD + kOff);
    bf16x8 hi = *(const bf16x8*)(base + row * KPAD + 16 + kOff);
    v16bf f;
#pragma unroll
    for (int q = 0; q < 8; ++q) { f[q] = lo[q]; f[q + 8] = hi[q]; }
    return f;
}

// ---------------------------------------------------------------------------
// GEMM1: act[row, j] = glu(gather(h)[row,:] @ gate_up[e][:, 2j / 2j+1] + bias)
// Block tile: 128 rows x 64 j-cols (=128 gate_up cols), K-step 32, double-buf.
// 8 waves in 4(M) x 2(N); wave = 32 rows x 32 j-cols = 2x2x{gate,up} WMMA accs.
// B staging: one column per thread-half -> paired cvt_pk + 2x ds_store_b128.
// ---------------------------------------------------------------------------
__global__ __launch_bounds__(256)
void gemm1_kernel(const __bf16* __restrict__ hb,
                  const float* __restrict__ gup,
                  const float* __restrict__ gub,
                  const int*  __restrict__ cnt,
                  const int*  __restrict__ off,
                  const int*  __restrict__ rows,
                  __bf16* __restrict__ act) {
    __shared__ __attribute__((aligned(16))) __bf16 As[2][128 * KPAD];
    __shared__ __attribute__((aligned(16))) __bf16 Bs[2][128 * KPAD];

    const int e  = blockIdx.x >> 5;
    const int mt = blockIdx.x & 31;
    const int Me = cnt[e];
    if (mt * 128 >= Me) return;               // block-uniform: EXEC stays full
    const int base  = off[e];
    const int jBase = blockIdx.y * 64;        // activation column tile
    const size_t gupBase = (size_t)e * H * TWOI + (size_t)(jBase * 2);

    const int tid  = threadIdx.x;
    const int lane = tid & 31;
    const int wave = tid >> 5;
    const int wm   = wave & 3;
    const int wn   = wave >> 2;

    // A-loader assignment: 128 rows x 4 k-groups of 8 halves, 2 segs/thread
    const int ar  = tid >> 2;                 // 0..63
    const int akg = tid & 3;                  // 0..3
    const int m0 = mt * 128 + ar;
    const int m1 = m0 + 64;
    const __bf16* aptr0 = hb + (size_t)rows[base + (m0 < Me ? m0 : 0)] * H;
    const __bf16* aptr1 = hb + (size_t)rows[base + (m1 < Me ? m1 : 0)] * H;

    // B-loader assignment: one gate_up column per thread-half
    const int bc  = tid & 127;                // column within 128-wide tile
    const int bkb = (tid >> 7) * 16;          // k-offset 0 or 16

    auto stageA = [&](int k0, int buf) {
#if HAVE_ASYNC_COPY
        async_copy16(aptr0 + k0 + akg * 8, As[buf] + ar * KPAD + akg * 8);
        async_copy16(aptr1 + k0 + akg * 8, As[buf] + (ar + 64) * KPAD + akg * 8);
#else
        *(bf16x8*)(As[buf] + ar * KPAD + akg * 8) =
            *(const bf16x8*)(aptr0 + k0 + akg * 8);
        *(bf16x8*)(As[buf] + (ar + 64) * KPAD + akg * 8) =
            *(const bf16x8*)(aptr1 + k0 + akg * 8);
#endif
    };
    auto stageB = [&](int k0, int buf) {
        const float* src = gup + gupBase + bc + (size_t)(k0 + bkb) * TWOI;
        float f[16];
#pragma unroll
        for (int i = 0; i < 16; ++i) f[i] = src[(size_t)i * TWOI];
        bf16x8 p0, p1;
#pragma unroll
        for (int i = 0; i < 8; ++i) { p0[i] = (__bf16)f[i]; p1[i] = (__bf16)f[8 + i]; }
        *(bf16x8*)(Bs[buf] + bc * KPAD + bkb)     = p0;   // 16B aligned
        *(bf16x8*)(Bs[buf] + bc * KPAD + bkb + 8) = p1;
    };

    v8f acc[2][2][2];
#pragma unroll
    for (int a = 0; a < 2; ++a)
#pragma unroll
        for (int b = 0; b < 2; ++b)
#pragma unroll
            for (int c = 0; c < 2; ++c) acc[a][b][c] = (v8f)0.0f;

    const int kOff = (lane < 16) ? 0 : 8;
    const int rsel = lane & 15;

    stageA(0, 0);
    stageB(0, 0);
#if HAVE_ASYNC_COPY
    async_wait0();
#endif
    __syncthreads();

    for (int s = 0; s < KSTEPS; ++s) {
        const int cur = s & 1;
        // 1) fragment loads from the current LDS buffer (issue DS reads first)
        v16bf a0 = make_frag(As[cur], wm * 32 + rsel,      kOff);
        v16bf a1 = make_frag(As[cur], wm * 32 + 16 + rsel, kOff);
        v16bf bg0 = make_frag(Bs[cur], 2 * (wn * 32 + rsel),          kOff);
        v16bf bu0 = make_frag(Bs[cur], 2 * (wn * 32 + rsel) + 1,      kOff);
        v16bf bg1 = make_frag(Bs[cur], 2 * (wn * 32 + 16 + rsel),     kOff);
        v16bf bu1 = make_frag(Bs[cur], 2 * (wn * 32 + 16 + rsel) + 1, kOff);
        // 2) stage the next tile into the other buffer
        if (s + 1 < KSTEPS) {
            stageA((s + 1) * BK, cur ^ 1);
            stageB((s + 1) * BK, cur ^ 1);
        }
        // 3) prefetch the fp32 weight tile two steps ahead (global_prefetch_b8)
        if (s + 2 < KSTEPS) {
            int kk = tid >> 3, c32 = (tid & 7) * 16;
            __builtin_prefetch(gup + gupBase + (size_t)((s + 2) * BK + kk) * TWOI + c32, 0, 1);
        }
        // 4) matrix math
        acc[0][0][0] = __builtin_amdgcn_wmma_f32_16x16x32_bf16(false, a0, false, bg0, (short)0, acc[0][0][0], false, false);
        acc[0][0][1] = __builtin_amdgcn_wmma_f32_16x16x32_bf16(false, a0, false, bu0, (short)0, acc[0][0][1], false, false);
        acc[0][1][0] = __builtin_amdgcn_wmma_f32_16x16x32_bf16(false, a0, false, bg1, (short)0, acc[0][1][0], false, false);
        acc[0][1][1] = __builtin_amdgcn_wmma_f32_16x16x32_bf16(false, a0, false, bu1, (short)0, acc[0][1][1], false, false);
        acc[1][0][0] = __builtin_amdgcn_wmma_f32_16x16x32_bf16(false, a1, false, bg0, (short)0, acc[1][0][0], false, false);
        acc[1][0][1] = __builtin_amdgcn_wmma_f32_16x16x32_bf16(false, a1, false, bu0, (short)0, acc[1][0][1], false, false);
        acc[1][1][0] = __builtin_amdgcn_wmma_f32_16x16x32_bf16(false, a1, false, bg1, (short)0, acc[1][1][0], false, false);
        acc[1][1][1] = __builtin_amdgcn_wmma_f32_16x16x32_bf16(false, a1, false, bu1, (short)0, acc[1][1][1], false, false);
        // 5) make next buffer visible
#if HAVE_ASYNC_COPY
        if (s + 1 < KSTEPS) async_wait0();
#endif
        __syncthreads();
    }

    // fused bias + clamp + SiLU-GLU epilogue, store bf16 activations
    const int rhi = (lane < 16) ? 0 : 8;
#pragma unroll
    for (int mi = 0; mi < 2; ++mi) {
#pragma unroll
        for (int ji = 0; ji < 2; ++ji) {
            int j  = jBase + wn * 32 + ji * 16 + rsel;
            float bgv = gub[(size_t)e * TWOI + 2 * j];
            float buv = gub[(size_t)e * TWOI + 2 * j + 1];
#pragma unroll
            for (int v = 0; v < 8; ++v) {
                int m = mt * 128 + wm * 32 + mi * 16 + v + rhi;
                if (m < Me) {
                    float g = acc[mi][ji][0][v] + bgv;
                    float u = acc[mi][ji][1][v] + buv;
                    g = fminf(g, LIMIT_C);
                    u = fminf(fmaxf(u, -LIMIT_C), LIMIT_C);
                    float sig = __builtin_amdgcn_rcpf(1.0f + __expf(-ALPHA_C * g));
                    float av  = (u + 1.0f) * (g * sig);
                    act[(size_t)(base + m) * I_DIM + j] = (__bf16)av;
                }
            }
        }
    }
}

// ---------------------------------------------------------------------------
// GEMM2: out[token, :] += score * (act[row,:] @ down[e] + down_bias[e])
// Block tile: 128 rows x 128 cols, K-step 32, double-buffered LDS.
// 8 waves in 4(M) x 2(N); wave = 32 rows x 64 cols = 2x4 WMMA accumulators.
// ---------------------------------------------------------------------------
__global__ __launch_bounds__(256)
void gemm2_kernel(const __bf16* __restrict__ act,
                  const float* __restrict__ dw,
                  const float* __restrict__ db,
                  const int*  __restrict__ cnt,
                  const int*  __restrict__ off,
                  const int*  __restrict__ rows,
                  const float* __restrict__ rscore,
                  float* __restrict__ out) {
    __shared__ __attribute__((aligned(16))) __bf16 As[2][128 * KPAD];
    __shared__ __attribute__((aligned(16))) __bf16 Bs[2][128 * KPAD];

    const int e  = blockIdx.x >> 5;
    const int mt = blockIdx.x & 31;
    const int Me = cnt[e];
    if (mt * 128 >= Me) return;
    const int base  = off[e];
    const int nBase = blockIdx.y * 128;
    const size_t dwBase = (size_t)e * I_DIM * H + nBase;

    const int tid  = threadIdx.x;
    const int lane = tid & 31;
    const int wave = tid >> 5;
    const int wm   = wave & 3;
    const int wn   = wave >> 2;

    const int ar  = tid >> 2;
    const int akg = tid & 3;
    // act rows are compacted; buffer has 128-row slack so no clamp needed
    const __bf16* aptr0 = act + (size_t)(base + mt * 128 + ar) * I_DIM;
    const __bf16* aptr1 = act + (size_t)(base + mt * 128 + ar + 64) * I_DIM;

    const int bc  = tid & 127;
    const int bkb = (tid >> 7) * 16;

    auto stageA = [&](int k0, int buf) {
#if HAVE_ASYNC_COPY
        async_copy16(aptr0 + k0 + akg * 8, As[buf] + ar * KPAD + akg * 8);
        async_copy16(aptr1 + k0 + akg * 8, As[buf] + (ar + 64) * KPAD + akg * 8);
#else
        *(bf16x8*)(As[buf] + ar * KPAD + akg * 8) =
            *(const bf16x8*)(aptr0 + k0 + akg * 8);
        *(bf16x8*)(As[buf] + (ar + 64) * KPAD + akg * 8) =
            *(const bf16x8*)(aptr1 + k0 + akg * 8);
#endif
    };
    auto stageB = [&](int k0, int buf) {
        const float* src = dw + dwBase + bc + (size_t)(k0 + bkb) * H;
        float f[16];
#pragma unroll
        for (int i = 0; i < 16; ++i) f[i] = src[(size_t)i * H];
        bf16x8 p0, p1;
#pragma unroll
        for (int i = 0; i < 8; ++i) { p0[i] = (__bf16)f[i]; p1[i] = (__bf16)f[8 + i]; }
        *(bf16x8*)(Bs[buf] + bc * KPAD + bkb)     = p0;
        *(bf16x8*)(Bs[buf] + bc * KPAD + bkb + 8) = p1;
    };

    v8f acc[2][4];
#pragma unroll
    for (int a = 0; a < 2; ++a)
#pragma unroll
        for (int b = 0; b < 4; ++b) acc[a][b] = (v8f)0.0f;

    const int kOff = (lane < 16) ? 0 : 8;
    const int rsel = lane & 15;

    stageA(0, 0);
    stageB(0, 0);
#if HAVE_ASYNC_COPY
    async_wait0();
#endif
    __syncthreads();

    for (int s = 0; s < KSTEPS; ++s) {
        const int cur = s & 1;
        v16bf a0 = make_frag(As[cur], wm * 32 + rsel,      kOff);
        v16bf a1 = make_frag(As[cur], wm * 32 + 16 + rsel, kOff);
        v16bf b0 = make_frag(Bs[cur], wn * 64 + rsel,      kOff);
        v16bf b1 = make_frag(Bs[cur], wn * 64 + 16 + rsel, kOff);
        v16bf b2 = make_frag(Bs[cur], wn * 64 + 32 + rsel, kOff);
        v16bf b3 = make_frag(Bs[cur], wn * 64 + 48 + rsel, kOff);
        if (s + 1 < KSTEPS) {
            stageA((s + 1) * BK, cur ^ 1);
            stageB((s + 1) * BK, cur ^ 1);
        }
        if (s + 2 < KSTEPS) {
            int kk = tid >> 3, c32 = (tid & 7) * 16;
            __builtin_prefetch(dw + dwBase + (size_t)((s + 2) * BK + kk) * H + c32, 0, 1);
        }
        acc[0][0] = __builtin_amdgcn_wmma_f32_16x16x32_bf16(false, a0, false, b0, (short)0, acc[0][0], false, false);
        acc[0][1] = __builtin_amdgcn_wmma_f32_16x16x32_bf16(false, a0, false, b1, (short)0, acc[0][1], false, false);
        acc[0][2] = __builtin_amdgcn_wmma_f32_16x16x32_bf16(false, a0, false, b2, (short)0, acc[0][2], false, false);
        acc[0][3] = __builtin_amdgcn_wmma_f32_16x16x32_bf16(false, a0, false, b3, (short)0, acc[0][3], false, false);
        acc[1][0] = __builtin_amdgcn_wmma_f32_16x16x32_bf16(false, a1, false, b0, (short)0, acc[1][0], false, false);
        acc[1][1] = __builtin_amdgcn_wmma_f32_16x16x32_bf16(false, a1, false, b1, (short)0, acc[1][1], false, false);
        acc[1][2] = __builtin_amdgcn_wmma_f32_16x16x32_bf16(false, a1, false, b2, (short)0, acc[1][2], false, false);
        acc[1][3] = __builtin_amdgcn_wmma_f32_16x16x32_bf16(false, a1, false, b3, (short)0, acc[1][3], false, false);
#if HAVE_ASYNC_COPY
        if (s + 1 < KSTEPS) async_wait0();
#endif
        __syncthreads();
    }

    const int rhi = (lane < 16) ? 0 : 8;
#pragma unroll
    for (int mi = 0; mi < 2; ++mi) {
#pragma unroll
        for (int ni = 0; ni < 4; ++ni) {
            int hcol = nBase + wn * 64 + ni * 16 + rsel;
            float bias = db[(size_t)e * H + hcol];
#pragma unroll
            for (int v = 0; v < 8; ++v) {
                int m = mt * 128 + wm * 32 + mi * 16 + v + rhi;
                if (m < Me) {
                    int   r  = base + m;
                    int   tk = rows[r];
                    float sc = rscore[r];
                    float val = (acc[mi][ni][v] + bias) * sc;
                    unsafeAtomicAdd(out + (size_t)tk * H + hcol, val);
                }
            }
        }
    }
}

// ---------------------------------------------------------------------------
// Host launcher
// ---------------------------------------------------------------------------
extern "C" void kernel_launch(void* const* d_in, const int* in_sizes, int n_in,
                              void* d_out, int out_size, void* d_ws, size_t ws_size,
                              hipStream_t stream) {
    (void)in_sizes; (void)n_in; (void)out_size; (void)ws_size;
    const float* h   = (const float*)d_in[0];
    const float* rw  = (const float*)d_in[1];
    const float* rb  = (const float*)d_in[2];
    const float* gup = (const float*)d_in[3];
    const float* gub = (const float*)d_in[4];
    const float* dw  = (const float*)d_in[5];
    const float* db  = (const float*)d_in[6];
    float* out = (float*)d_out;

    char*  wsb = (char*)d_ws;
    size_t pos = 0;
    auto alloc = [&](size_t bytes) -> void* {
        void* r = (void*)(wsb + pos);
        pos = (pos + bytes + 255) & ~(size_t)255;
        return r;
    };
    __bf16* hb     = (__bf16*)alloc((size_t)T * H * 2);
    __bf16* act    = (__bf16*)alloc((size_t)(PAIRS + 128) * I_DIM * 2);
    int*    rowsv  = (int*)   alloc((size_t)(PAIRS + 128) * 4);
    float*  rscore = (float*) alloc((size_t)(PAIRS + 128) * 4);
    int*    cnt    = (int*)   alloc(E * 4);
    int*    off    = (int*)   alloc((E + 1) * 4);
    int*    fill   = (int*)   alloc(E * 4);
    int*    topidx = (int*)   alloc((size_t)T * 2 * 4);
    float*  topw   = (float*) alloc((size_t)T * 2 * 4);

    (void)hipMemsetAsync(out, 0, (size_t)T * H * sizeof(float), stream);
    (void)hipMemsetAsync(cnt, 0, E * sizeof(int), stream);

    cvt_hidden_kernel<<<(T * H) / 1024, 256, 0, stream>>>(h, hb);
    router_kernel<<<T / 8, 256, 0, stream>>>(h, rw, rb, cnt, topidx, topw);
    offsets_kernel<<<1, 32, 0, stream>>>(cnt, off, fill);
    scatter_kernel<<<T / 256, 256, 0, stream>>>(topidx, topw, off, fill, rowsv, rscore);

    dim3 g1(E * 32, I_DIM / 64);
    gemm1_kernel<<<g1, 256, 0, stream>>>(hb, gup, gub, cnt, off, rowsv, act);

    dim3 g2(E * 32, H / 128);
    gemm2_kernel<<<g2, 256, 0, stream>>>(act, dw, db, cnt, off, rowsv, rscore, out);
}
